// MultiHeadSelfAttention_13391708029451
// MI455X (gfx1250) — matrix operand order repaired and checked by reading, not compiled, not run on hardware
//
#include <hip/hip_runtime.h>

#define EMBED  1024
#define SEQ    2048
#define NHEAD  16
#define HD     64
#define TOKENS 4096   // BATCH * SEQ

typedef __attribute__((ext_vector_type(16))) _Float16 v16h;
typedef __attribute__((ext_vector_type(8)))  float    v8f;

union Frag16 { v16h v; uint4 u[2]; };

__device__ __forceinline__ v8f wmma_f16(v16h a, v16h b, v8f c) {
  // D = A(16x32 f16) * B(32x16 f16) + C(16x16 f32)
  return __builtin_amdgcn_wmma_f32_16x16x32_f16(false, a, false, b, (short)0, c,
                                                false, false);
}

// CDNA5 async global->LDS copy (ASYNCcnt-tracked, no VGPR staging).
// INST_OFFSET is added to BOTH the LDS and the global address, so callers pass
// an LDS base compensated by -OFF and share one global base VGPR pair across a
// whole tile of asyncs (no per-instruction address VALU).
template <int OFF>
__device__ __forceinline__ void async_b128o(unsigned lds, const _Float16* g) {
  asm volatile("global_load_async_to_lds_b128 %0, %1, off offset:%c2"
               :: "v"(lds), "v"(g), "n"(OFF) : "memory");
}
__device__ __forceinline__ void wait_async0() {
  asm volatile("s_wait_asynccnt 0x0" ::: "memory");
}

// Paired K/V tile issue for the flash kernel: 4+4 b128 asyncs, all immediate.
template <int KO, int VO>
__device__ __forceinline__ void issue_kv_t(unsigned lK, unsigned lV,
                                           const _Float16* gK,
                                           const _Float16* gV) {
  async_b128o<KO + 0>(lK, gK);  async_b128o<KO + 16>(lK, gK);
  async_b128o<KO + 32>(lK, gK); async_b128o<KO + 48>(lK, gK);
  async_b128o<VO + 0>(lV, gV);  async_b128o<VO + 16>(lV, gV);
  async_b128o<VO + 32>(lV, gV); async_b128o<VO + 48>(lV, gV);
}

// ---------------------------------------------------------------------------
// fp32 -> f16 elementwise (vectorized)
// ---------------------------------------------------------------------------
__global__ void __launch_bounds__(256) k_cvt(const float* __restrict__ in,
                                             _Float16* __restrict__ out, int n4) {
  int i = blockIdx.x * blockDim.x + threadIdx.x;
  if (i >= n4) return;
  float4 f = ((const float4*)in)[i];
  union { _Float16 h[4]; uint2 u; } p;
  p.h[0] = (_Float16)f.x; p.h[1] = (_Float16)f.y;
  p.h[2] = (_Float16)f.z; p.h[3] = (_Float16)f.w;
  ((uint2*)out)[i] = p.u;
}

// ---------------------------------------------------------------------------
// fp32 [K][N] -> f16 transposed [N][K]   (EMBED x EMBED, LDS tile transpose)
// ---------------------------------------------------------------------------
__global__ void __launch_bounds__(256) k_cvtT(const float* __restrict__ w,
                                              _Float16* __restrict__ wT) {
  __shared__ _Float16 tile[32][33];
  const int nt = blockIdx.x, kt = blockIdx.y;
  const int x = threadIdx.x, y0 = threadIdx.y * 4;
#pragma unroll
  for (int i = 0; i < 4; ++i)
    tile[y0 + i][x] = (_Float16)w[(size_t)(kt * 32 + y0 + i) * EMBED + nt * 32 + x];
  __syncthreads();
#pragma unroll
  for (int i = 0; i < 4; ++i)
    wT[(size_t)(nt * 32 + y0 + i) * EMBED + kt * 32 + x] = tile[x][y0 + i];
}

// ---------------------------------------------------------------------------
// V [token][EMBED] f16 -> VT [bh][HD][SEQ] f16 (per-head transpose so the
// flash kernel can async-copy V tiles directly in B^t ([d][key]) layout)
// ---------------------------------------------------------------------------
__global__ void __launch_bounds__(256) k_vT(const _Float16* __restrict__ Vh,
                                            _Float16* __restrict__ VT) {
  __shared__ alignas(16) _Float16 tile[64][72];
  const int st = blockIdx.x;            // seq tile of 64
  const int bh = blockIdx.y;
  const int b = bh >> 4, h = bh & 15;
  const int t = threadIdx.x;
  const int r = t >> 2, c0 = (t & 3) * 16;
  const uint4* g = (const uint4*)(Vh + ((size_t)b * SEQ + st * 64 + r) * EMBED +
                                  h * HD + c0);
  *(uint4*)(&tile[r][c0])     = g[0];
  *(uint4*)(&tile[r][c0 + 8]) = g[1];
  __syncthreads();
  union { uint4 u; _Float16 h8[8]; } o0, o1;
#pragma unroll
  for (int e = 0; e < 8; ++e) {
    o0.h8[e] = tile[c0 + e][r];
    o1.h8[e] = tile[c0 + 8 + e][r];
  }
  _Float16* o = VT + ((size_t)bh * HD + r) * SEQ + st * 64 + c0;
  *(uint4*)o       = o0.u;
  *((uint4*)o + 1) = o1.u;
}

// ---------------------------------------------------------------------------
// WMMA GEMM body: C[128x128] = X[128xK] * WT[128xK]^T, K = EMBED.
// 8 waves, wave (wm,wn) owns 32x64 (2x4 frags). Double-buffered LDS tiles
// filled by async-to-LDS with immediate offsets; branch-free unroll-2 loop.
// Fragments preloaded before the 8-WMMA burst.
// ---------------------------------------------------------------------------
__device__ __forceinline__ void gemm_block(const _Float16* __restrict__ X,
                                           const _Float16* __restrict__ WT,
                                           int m0, int n0,
                                           _Float16* As, _Float16* Bs,
                                           v8f acc[2][4]) {
  const int t = threadIdx.x;
  const int lane = t & 31, wvid = t >> 5;
  const int wm = wvid & 3, wn = wvid >> 2;
  const int half = lane >> 4, lr = lane & 15;
  const int r = t >> 1, c = (t & 1) * 16;

  const _Float16* pa = X + (size_t)(m0 + r) * EMBED + c;   // tile-kt base
  const _Float16* pb = WT + (size_t)(n0 + r) * EMBED + c;
  const unsigned lA0 = (unsigned)(size_t)(As + r * 40 + c);
  const unsigned lB0 = (unsigned)(size_t)(Bs + r * 40 + c);
  const unsigned lA1c = lA0 + 10240u - 64u;    // buf1, compensated for offset:64
  const unsigned lB1c = lB0 + 10240u - 64u;
  const unsigned lA0c = lA0 - 128u;            // buf0, compensated for offset:128
  const unsigned lB0c = lB0 - 128u;

  const int arow0 = wm * 32 + lr;
  const int bcol0 = wn * 64 + lr;

  auto compute = [&](int buf) {
    const _Float16* Ab = As + buf * 5120;
    const _Float16* Bb = Bs + buf * 5120;
    Frag16 af[2], bf[4];
#pragma unroll
    for (int mi = 0; mi < 2; ++mi) {
      const int row = arow0 + mi * 16;
      // A 16x32 f16 layout: lanes<16 K=0-7,16-23 ; lanes>=16 K=8-15,24-31
      af[mi].u[0] = *(const uint4*)(Ab + row * 40 + half * 8);
      af[mi].u[1] = *(const uint4*)(Ab + row * 40 + half * 8 + 16);
    }
#pragma unroll
    for (int ni = 0; ni < 4; ++ni) {
      const int col = bcol0 + ni * 16;
      // B 32x16 f16 from [N][K] tile: lanes<16 K=0-7,16-23 ; lanes>=16 K=8-15,24-31
      bf[ni].u[0] = *(const uint4*)(Bb + col * 40 + half * 8);
      bf[ni].u[1] = *(const uint4*)(Bb + col * 40 + half * 8 + 16);
    }
#pragma unroll
    for (int ni = 0; ni < 4; ++ni) {
      acc[0][ni] = wmma_f16(af[0].v, bf[ni].v, acc[0][ni]);
      acc[1][ni] = wmma_f16(af[1].v, bf[ni].v, acc[1][ni]);
    }
  };

  auto sync = [&]() { wait_async0(); __syncthreads(); };

  // tile 0 -> buf0
  async_b128o<0>(lA0, pa);  async_b128o<16>(lA0, pa);
  async_b128o<0>(lB0, pb);  async_b128o<16>(lB0, pb);
  sync();
  for (int kt = 0; kt < EMBED / 32 - 2; kt += 2) {   // branch-free body
    async_b128o<64>(lA1c, pa);   async_b128o<80>(lA1c, pa);   // tile kt+1 -> buf1
    async_b128o<64>(lB1c, pb);   async_b128o<80>(lB1c, pb);
    compute(0);
    sync();
    async_b128o<128>(lA0c, pa);  async_b128o<144>(lA0c, pa);  // tile kt+2 -> buf0
    async_b128o<128>(lB0c, pb);  async_b128o<144>(lB0c, pb);
    compute(1);
    sync();
    pa += 64;  pb += 64;         // advance 2 tiles (128 bytes)
  }
  async_b128o<64>(lA1c, pa);   async_b128o<80>(lA1c, pa);     // tile 31 -> buf1
  async_b128o<64>(lB1c, pb);   async_b128o<80>(lB1c, pb);
  compute(0);
  sync();
  compute(1);
}

// ---------------------------------------------------------------------------
// Fused Q/K/V projection: blockIdx.z selects weight/bias/output. f16 output.
// ---------------------------------------------------------------------------
__global__ void __launch_bounds__(256) k_gemm_qkv(
    const _Float16* __restrict__ X, const _Float16* __restrict__ WT3,
    const float* __restrict__ bq, const float* __restrict__ bk,
    const float* __restrict__ bv, _Float16* __restrict__ OUT3) {
  __shared__ alignas(16) _Float16 As[2 * 5120];
  __shared__ alignas(16) _Float16 Bs[2 * 5120];
  const int z = blockIdx.z;
  const _Float16* WT = WT3 + (size_t)z * EMBED * EMBED;
  const float* bias = (z == 0) ? bq : (z == 1) ? bk : bv;
  _Float16* out = OUT3 + (size_t)z * TOKENS * EMBED;
  const int m0 = blockIdx.y * 128, n0 = blockIdx.x * 128;

  v8f acc[2][4] = {};
  gemm_block(X, WT, m0, n0, As, Bs, acc);

  const int lane = threadIdx.x & 31, wvid = threadIdx.x >> 5;
  const int wm = wvid & 3, wn = wvid >> 2;
  const int half = lane >> 4, lr = lane & 15;
#pragma unroll
  for (int mi = 0; mi < 2; ++mi)
#pragma unroll
    for (int ni = 0; ni < 4; ++ni) {
      const int col = n0 + wn * 64 + ni * 16 + lr;
      const float bcol = bias[col];
#pragma unroll
      for (int rr = 0; rr < 8; ++rr) {
        const int row = m0 + wm * 32 + mi * 16 + rr + 8 * half;
        out[(size_t)row * EMBED + col] = (_Float16)(acc[mi][ni][rr] + bcol);
      }
    }
}

// ---------------------------------------------------------------------------
// Output projection: ctx(f16) @ woT + bo -> f32 d_out
// ---------------------------------------------------------------------------
__global__ void __launch_bounds__(256) k_gemm_out(
    const _Float16* __restrict__ X, const _Float16* __restrict__ WT,
    const float* __restrict__ bias, float* __restrict__ out) {
  __shared__ alignas(16) _Float16 As[2 * 5120];
  __shared__ alignas(16) _Float16 Bs[2 * 5120];
  const int m0 = blockIdx.y * 128, n0 = blockIdx.x * 128;

  v8f acc[2][4] = {};
  gemm_block(X, WT, m0, n0, As, Bs, acc);

  const int lane = threadIdx.x & 31, wvid = threadIdx.x >> 5;
  const int wm = wvid & 3, wn = wvid >> 2;
  const int half = lane >> 4, lr = lane & 15;
#pragma unroll
  for (int mi = 0; mi < 2; ++mi)
#pragma unroll
    for (int ni = 0; ni < 4; ++ni) {
      const int col = n0 + wn * 64 + ni * 16 + lr;
      const float bcol = bias[col];
#pragma unroll
      for (int rr = 0; rr < 8; ++rr) {
        const int row = m0 + wm * 32 + mi * 16 + rr + 8 * half;
        out[(size_t)row * EMBED + col] = acc[mi][ni][rr] + bcol;
      }
    }
}

// ---------------------------------------------------------------------------
// Flash attention: block = 64 queries of one (batch, head). 128 threads.
// Wave w owns query rows [16w,16w+16) and the whole 64-key tile.
// K and V^T tiles double-buffered via immediate-offset async-to-LDS;
// branch-free unroll-2 body; fragments preloaded before each WMMA burst.
// ---------------------------------------------------------------------------
__global__ void __launch_bounds__(128) k_attn(const _Float16* __restrict__ Q,
                                              const _Float16* __restrict__ K,
                                              const _Float16* __restrict__ VT,
                                              _Float16* __restrict__ CTX) {
  __shared__ alignas(16) _Float16 Qs[64 * 72];
  __shared__ alignas(16) _Float16 Ks[2][64 * 72];   // [key][d]  == B^t for Q*K^T
  __shared__ alignas(16) _Float16 Vs[2][64 * 72];   // [d][key]  == B^t for P*V
  __shared__ alignas(16) _Float16 Ps[4][16 * 72];   // per-wave P tile [q][key]

  const int t = threadIdx.x, lane = t & 31, w = t >> 5;
  const int half = lane >> 4, lr = lane & 15;
  const int q0 = blockIdx.x * 64;
  const int bh = blockIdx.y;
  const size_t rowbase = (size_t)(bh >> 4) * SEQ;
  const int cbase = (bh & 15) * HD;
  const float scale = 0.125f;           // 1/sqrt(64)

  const int r = t >> 1, c0 = (t & 1) * 32;

  // tile strides in bytes: K tile = 64 rows * EMBED * 2 = 131072; V tile = 128
  const _Float16* gK = K + (rowbase + r) * EMBED + cbase + c0;        // tile 0
  const _Float16* gV = VT + ((size_t)bh * HD + r) * SEQ + c0;         // tile 0
  const unsigned lK0 = (unsigned)(size_t)(&Ks[0][r * 72 + c0]);
  const unsigned lV0 = (unsigned)(size_t)(&Vs[0][r * 72 + c0]);
  const unsigned lK1c = lK0 + 9216u - 131072u;   // buf1, comp. for offset:131072
  const unsigned lV1c = lV0 + 9216u - 128u;      // buf1, comp. for offset:128
  const unsigned lK0c = lK0 - 262144u;           // buf0, comp. for offset:262144
  const unsigned lV0c = lV0 - 256u;              // buf0, comp. for offset:256

  // prologue: Q tile + first K/V tiles, all async
  {
    const _Float16* gQ = Q + (rowbase + q0 + r) * EMBED + cbase + c0;
    const unsigned lQ = (unsigned)(size_t)(Qs + r * 72 + c0);
    async_b128o<0>(lQ, gQ);  async_b128o<16>(lQ, gQ);
    async_b128o<32>(lQ, gQ); async_b128o<48>(lQ, gQ);
  }
  issue_kv_t<0, 0>(lK0, lV0, gK, gV);
  wait_async0();
  __syncthreads();

  Frag16 qf[2];
#pragma unroll
  for (int ks = 0; ks < 2; ++ks) {
    const int row = w * 16 + lr;
    qf[ks].u[0] = *(const uint4*)(Qs + row * 72 + 32 * ks + half * 8);
    qf[ks].u[1] = *(const uint4*)(Qs + row * 72 + 32 * ks + half * 8 + 16);
  }

  v8f acc[4] = {};
  float mrow[8], lrow[8];
#pragma unroll
  for (int i = 0; i < 8; ++i) { mrow[i] = -1e30f; lrow[i] = 0.0f; }

  _Float16* Pw = Ps[w];

  auto tile_compute = [&](int buf) {
    // ---- scores S = Q * K^T ----
    const _Float16* Kb = Ks[buf];
    v8f sc[4] = {};
#pragma unroll
    for (int ks = 0; ks < 2; ++ks) {
      Frag16 bf[4];
#pragma unroll
      for (int ni = 0; ni < 4; ++ni) {
        const int col = ni * 16 + lr;
        bf[ni].u[0] = *(const uint4*)(Kb + col * 72 + 32 * ks + half * 8);
        bf[ni].u[1] = *(const uint4*)(Kb + col * 72 + 32 * ks + half * 8 + 16);
      }
#pragma unroll
      for (int ni = 0; ni < 4; ++ni)
        sc[ni] = wmma_f16(qf[ks].v, bf[ni].v, sc[ni]);
    }

    // ---- online softmax (row = VGPR slot + half; 16 cols across 16 lanes) ----
#pragma unroll
    for (int r8 = 0; r8 < 8; ++r8) {
      float s0 = sc[0][r8] * scale, s1 = sc[1][r8] * scale;
      float s2 = sc[2][r8] * scale, s3 = sc[3][r8] * scale;
      float rmax = fmaxf(fmaxf(s0, s1), fmaxf(s2, s3));
#pragma unroll
      for (int m = 1; m < 16; m <<= 1) rmax = fmaxf(rmax, __shfl_xor(rmax, m, 32));
      const float mnew  = fmaxf(mrow[r8], rmax);
      const float alpha = __expf(mrow[r8] - mnew);
      const float p0 = __expf(s0 - mnew), p1 = __expf(s1 - mnew);
      const float p2 = __expf(s2 - mnew), p3 = __expf(s3 - mnew);
      float psum = (p0 + p1) + (p2 + p3);
#pragma unroll
      for (int m = 1; m < 16; m <<= 1) psum += __shfl_xor(psum, m, 32);
      lrow[r8] = lrow[r8] * alpha + psum;
      mrow[r8] = mnew;
      acc[0][r8] *= alpha; acc[1][r8] *= alpha;
      acc[2][r8] *= alpha; acc[3][r8] *= alpha;
      const int prow = (r8 + 8 * half) * 72;
      Pw[prow + 0 * 16 + lr] = (_Float16)p0;
      Pw[prow + 1 * 16 + lr] = (_Float16)p1;
      Pw[prow + 2 * 16 + lr] = (_Float16)p2;
      Pw[prow + 3 * 16 + lr] = (_Float16)p3;
    }
    asm volatile("" ::: "memory");  // wave-private LDS: DS ops in-order per wave

    // ---- acc += P * V ----
    const _Float16* Vb = Vs[buf];
#pragma unroll
    for (int ks = 0; ks < 2; ++ks) {
      Frag16 pf, vf[4];
      pf.u[0] = *(const uint4*)(Pw + lr * 72 + 32 * ks + half * 8);
      pf.u[1] = *(const uint4*)(Pw + lr * 72 + 32 * ks + half * 8 + 16);
#pragma unroll
      for (int ni = 0; ni < 4; ++ni) {
        const int col = ni * 16 + lr;
        vf[ni].u[0] = *(const uint4*)(Vb + col * 72 + 32 * ks + half * 8);
        vf[ni].u[1] = *(const uint4*)(Vb + col * 72 + 32 * ks + half * 8 + 16);
      }
#pragma unroll
      for (int ni = 0; ni < 4; ++ni)
        acc[ni] = wmma_f16(pf.v, vf[ni].v, acc[ni]);
    }
  };

  auto sync = [&]() { wait_async0(); __syncthreads(); };

  for (int kt = 0; kt < SEQ / 64 - 2; kt += 2) {   // branch-free body
    issue_kv_t<131072, 128>(lK1c, lV1c, gK, gV);   // tile kt+1 -> buf1
    tile_compute(0);
    sync();
    issue_kv_t<262144, 256>(lK0c, lV0c, gK, gV);   // tile kt+2 -> buf0
    tile_compute(1);
    sync();
    gK += 131072;   // 2 K tiles (halves)
    gV += 128;      // 2 V tiles (halves)
  }
  issue_kv_t<131072, 128>(lK1c, lV1c, gK, gV);     // tile 31 -> buf1
  tile_compute(0);
  sync();
  tile_compute(1);

  // ---- normalize + store ctx (f16) ----
#pragma unroll
  for (int r8 = 0; r8 < 8; ++r8) {
    const float inv = 1.0f / lrow[r8];
    const int row = q0 + w * 16 + r8 + 8 * half;
#pragma unroll
    for (int ni = 0; ni < 4; ++ni)
      CTX[(rowbase + row) * EMBED + cbase + ni * 16 + lr] =
          (_Float16)(acc[ni][r8] * inv);
  }
}

// ---------------------------------------------------------------------------
extern "C" void kernel_launch(void* const* d_in, const int* in_sizes, int n_in,
                              void* d_out, int out_size, void* d_ws, size_t ws_size,
                              hipStream_t stream) {
  (void)in_sizes; (void)n_in; (void)out_size; (void)ws_size;
  const float* x  = (const float*)d_in[0];
  const float* wq = (const float*)d_in[1];
  const float* bq = (const float*)d_in[2];
  const float* wk = (const float*)d_in[3];
  const float* bk = (const float*)d_in[4];
  const float* wv = (const float*)d_in[5];
  const float* bv = (const float*)d_in[6];
  const float* wo = (const float*)d_in[7];
  const float* bo = (const float*)d_in[8];
  float* out = (float*)d_out;

  char* ws = (char*)d_ws;
  _Float16* xh   = (_Float16*)(ws);                 //  8 MB
  _Float16* wT3  = (_Float16*)(ws + 8388608);       //  6 MB (wqT,wkT,wvT)
  _Float16* woT  = (_Float16*)(ws + 14680064);      //  2 MB
  _Float16* qkv  = (_Float16*)(ws + 16777216);      // 24 MB (Q,K,V)
  _Float16* ctxh = (_Float16*)(ws + 41943040);      //  8 MB
  _Float16* vT   = (_Float16*)(ws + 50331648);      //  8 MB (V^T per head)
  _Float16* Qh = qkv;
  _Float16* Kh = qkv + (size_t)TOKENS * EMBED;
  _Float16* Vh = qkv + (size_t)2 * TOKENS * EMBED;

  // 1) convert activations + weights (weights transposed to [N][K])
  k_cvt<<<(TOKENS * EMBED / 4 + 255) / 256, 256, 0, stream>>>(x, xh, TOKENS * EMBED / 4);
  dim3 tb(32, 8);
  k_cvtT<<<dim3(32, 32), tb, 0, stream>>>(wq, wT3);
  k_cvtT<<<dim3(32, 32), tb, 0, stream>>>(wk, wT3 + (size_t)EMBED * EMBED);
  k_cvtT<<<dim3(32, 32), tb, 0, stream>>>(wv, wT3 + (size_t)2 * EMBED * EMBED);
  k_cvtT<<<dim3(32, 32), tb, 0, stream>>>(wo, woT);

  // 2) fused QKV projection (WMMA, async double-buffered)
  k_gemm_qkv<<<dim3(EMBED / 128, TOKENS / 128, 3), 256, 0, stream>>>(
      xh, wT3, bq, bk, bv, qkv);

  // 3) per-head V transpose for the flash kernel's B^t layout
  k_vT<<<dim3(SEQ / 64, 32), 256, 0, stream>>>(Vh, vT);

  // 4) flash attention (WMMA, async double-buffered K/V^T)
  k_attn<<<dim3(SEQ / 64, 32), 128, 0, stream>>>(Qh, Kh, vT, ctxh);

  // 5) output projection -> f32 (WMMA, async double-buffered)
  k_gemm_out<<<dim3(EMBED / 128, TOKENS / 128), 256, 0, stream>>>(ctxh, woT, bo, out);
}
